// BasicTransformer2D_32366873543457
// MI455X (gfx1250) — matrix operand order repaired and checked by reading, not compiled, not run on hardware
//
#include <hip/hip_runtime.h>
#include <hip/hip_bf16.h>
#include <cstdint>

typedef __attribute__((ext_vector_type(16))) _Float16 v16h;
typedef __attribute__((ext_vector_type(8)))  float    v8f;
typedef __attribute__((ext_vector_type(8)))  unsigned int v8u;

#define S_  4096
#define C_  256
#define NH_ 8
#define HD_ 32
#define LC_ 77
#define DC_ 768
#define FF_ 1024
#define NLAYER 2

__device__ __forceinline__ v16h u8_to_h16(v8u u) {
    union { v8u u; v16h h; } c; c.u = u; return c.h;
}

__device__ __forceinline__ float red_max16(float v) {
    v = fmaxf(v, __shfl_xor(v, 1, 32));
    v = fmaxf(v, __shfl_xor(v, 2, 32));
    v = fmaxf(v, __shfl_xor(v, 4, 32));
    v = fmaxf(v, __shfl_xor(v, 8, 32));
    return v;
}
__device__ __forceinline__ float red_sum16(float v) {
    v += __shfl_xor(v, 1, 32);
    v += __shfl_xor(v, 2, 32);
    v += __shfl_xor(v, 4, 32);
    v += __shfl_xor(v, 8, 32);
    return v;
}
__device__ __forceinline__ float red_max32(float v) { v = red_max16(v); return fmaxf(v, __shfl_xor(v, 16, 32)); }
__device__ __forceinline__ float red_sum32(float v) { v = red_sum16(v); return v + __shfl_xor(v, 16, 32); }

// A-fragment K index pattern (16-bit A 16x32, ISA 7.12.2)
__device__ __forceinline__ int a_kidx(int v, int hl) {
    return ((v < 4) ? 2 * v : 16 + 2 * (v - 4)) + hl * 8;
}

// ---------------------------------------------------------------- transposes
__global__ void k_transpose_in(const float* __restrict__ hid, float* __restrict__ x) {
    int i = blockIdx.x * blockDim.x + threadIdx.x;   // over S*C
    if (i >= S_ * C_) return;
    int s = i / C_, c = i % C_;
    x[i] = hid[(size_t)c * S_ + s];
}
__global__ void k_transpose_out(const float* __restrict__ x, float* __restrict__ out) {
    int i = blockIdx.x * blockDim.x + threadIdx.x;   // over C*S
    if (i >= S_ * C_) return;
    int c = i / S_, s = i % S_;
    out[i] = x[(size_t)s * C_ + c];
}
__global__ void k_f32_to_f16(const float* __restrict__ in, _Float16* __restrict__ out, int n) {
    int i = blockIdx.x * blockDim.x + threadIdx.x;
    for (; i < n; i += gridDim.x * blockDim.x) out[i] = (_Float16)in[i];
}

// ---------------------------------------------------------------- layernorm
__global__ __launch_bounds__(256) void k_layernorm(const float* __restrict__ x,
                                                   const float* __restrict__ g,
                                                   const float* __restrict__ b,
                                                   _Float16* __restrict__ out, int rows) {
    int lane = threadIdx.x & 31, wid = threadIdx.x >> 5;
    int row = blockIdx.x * 8 + wid;
    if (row >= rows) return;
    const float* xr = x + (size_t)row * C_;
    float v[8]; float s = 0.f;
#pragma unroll
    for (int i = 0; i < 8; i++) { v[i] = xr[lane + 32 * i]; s += v[i]; }
    s = red_sum32(s);
    float mean = s * (1.f / C_);
    float q = 0.f;
#pragma unroll
    for (int i = 0; i < 8; i++) { float d = v[i] - mean; q += d * d; }
    q = red_sum32(q);
    float rstd = rsqrtf(q * (1.f / C_) + 1e-5f);
    _Float16* orow = out + (size_t)row * C_;
#pragma unroll
    for (int i = 0; i < 8; i++) {
        int c = lane + 32 * i;
        orow[c] = (_Float16)((v[i] - mean) * rstd * g[c] + b[c]);
    }
}

// ---------------------------------------------------------------- tiled WMMA GEMM
// out[M,N] = A[M,K] @ B[K,N] (+bias +resid, opt GELU). 64x64 block tile, 4 waves.
// outHT (optional): f16 output stored transposed, outHT[n*M + m]  (ld = M).
__global__ __launch_bounds__(128) void k_gemm(const _Float16* __restrict__ A,
                                              const _Float16* __restrict__ Bw,
                                              const float* __restrict__ bias,
                                              const float* __restrict__ resid,
                                              float* __restrict__ outF,
                                              _Float16* __restrict__ outH,
                                              _Float16* __restrict__ outHT,
                                              int M, int N, int K, int dogelu) {
    __shared__ _Float16 sA[64 * 32];    // row-major [row][k]
    __shared__ _Float16 sBt[64 * 32];   // col-major [n][k]
    int tid = threadIdx.x;
    int lane = tid & 31, wid = tid >> 5;
    int wy = wid >> 1, wx = wid & 1;
    int hl = lane >> 4, l16 = lane & 15;
    int bm = blockIdx.y * 64, bn = blockIdx.x * 64;

    v8f acc[2][2] = {};

    for (int kb = 0; kb < K; kb += 32) {
        // ---- load A tile 64x32 (each thread: 16 contiguous f16)
        {
            int row = tid >> 1, hr = (tid & 1) * 16;
            int grow = bm + row;
            _Float16* dst = sA + row * 32 + hr;
            if (grow < M) {
                const _Float16* src = A + (size_t)grow * K + kb + hr;
                *(float4*)dst       = *(const float4*)src;
                *(float4*)(dst + 8) = *(const float4*)(src + 8);
            } else {
                *(float4*)dst = make_float4(0.f, 0.f, 0.f, 0.f);
                *(float4*)(dst + 8) = make_float4(0.f, 0.f, 0.f, 0.f);
            }
        }
        // ---- load B tile 32x64 transposed into sBt [n][k]
        {
            int k = tid >> 2, nseg = (tid & 3) * 16;
            const _Float16* src = Bw + (size_t)(kb + k) * N + bn + nseg;
            union { float4 f; _Float16 h[8]; } u;
#pragma unroll
            for (int half8 = 0; half8 < 2; half8++) {
                u.f = *(const float4*)(src + half8 * 8);
#pragma unroll
                for (int j = 0; j < 8; j++)
                    sBt[(nseg + half8 * 8 + j) * 32 + k] = u.h[j];
            }
        }
        __syncthreads();

        // ---- build fragments & WMMA
        v16h afr[2], bfr[2];
#pragma unroll
        for (int i = 0; i < 2; i++) {
            int rbase = wy * 32 + i * 16 + l16;
            v8u au;
#pragma unroll
            for (int v = 0; v < 8; v++)
                au[v] = *(const unsigned int*)(sA + rbase * 32 + a_kidx(v, hl));
            afr[i] = u8_to_h16(au);
        }
#pragma unroll
        for (int j = 0; j < 2; j++) {
            int cbase = wx * 32 + j * 16 + l16;
            v8u bu;
#pragma unroll
            for (int v = 0; v < 8; v++)
                bu[v] = *(const unsigned int*)(sBt + cbase * 32 + a_kidx(v, hl));
            bfr[j] = u8_to_h16(bu);
        }
#pragma unroll
        for (int i = 0; i < 2; i++)
#pragma unroll
            for (int j = 0; j < 2; j++)
                acc[i][j] = __builtin_amdgcn_wmma_f32_16x16x32_f16(
                    false, afr[i], false, bfr[j], (short)0, acc[i][j], false, false);
        __syncthreads();
    }

    // ---- epilogue
#pragma unroll
    for (int i = 0; i < 2; i++) {
#pragma unroll
        for (int j = 0; j < 2; j++) {
            int colg = bn + wx * 32 + j * 16 + l16;
#pragma unroll
            for (int v = 0; v < 8; v++) {
                int rowg = bm + wy * 32 + i * 16 + v + hl * 8;
                if (rowg >= M) continue;
                float val = acc[i][j][v];
                if (bias)  val += bias[colg];
                if (resid) val += resid[(size_t)rowg * N + colg];
                if (dogelu) val = 0.5f * val * (1.f + erff(val * 0.70710678118654752f));
                if (outF)  outF[(size_t)rowg * N + colg] = val;
                if (outH)  outH[(size_t)rowg * N + colg] = (_Float16)val;
                if (outHT) outHT[(size_t)colg * M + rowg] = (_Float16)val;
            }
        }
    }
}

// ---------------------------------------------------------------- flash self-attention
// grid (S/64, NH), 128 threads (4 waves); one wave = 16 queries of one head.
// Vt is the transposed V: Vt[(h*HD + d) * S + s]
__global__ __launch_bounds__(128) void k_flash_attn(const _Float16* __restrict__ Q,
                                                    const _Float16* __restrict__ Km,
                                                    const _Float16* __restrict__ Vt,
                                                    const float* __restrict__ tab,
                                                    _Float16* __restrict__ out) {
    __shared__ _Float16 sP[4][16 * 32];
    int lane = threadIdx.x & 31, wid = threadIdx.x >> 5;
    int hl = lane >> 4, l16 = lane & 15;
    int h = blockIdx.y;
    int qbase = blockIdx.x * 64 + wid * 16;
    const float scale = 0.17677669529663687f;  // 1/sqrt(32)

    // Q A-fragment, resident for whole kernel
    v8u qu;
    const _Float16* qp = Q + (size_t)(qbase + l16) * C_ + h * HD_;
#pragma unroll
    for (int v = 0; v < 8; v++)
        qu[v] = *(const unsigned int*)(qp + a_kidx(v, hl));
    v16h qf = u8_to_h16(qu);

    // per-lane base pointers for V^T fragments (column d = dbase + l16)
    const _Float16* vt0 = Vt + (size_t)(h * HD_ + l16) * S_;        // d in [0,16)
    const _Float16* vt1 = Vt + (size_t)(h * HD_ + 16 + l16) * S_;   // d in [16,32)

    v8f o0 = {}, o1 = {};
    float mrow[8], lrow[8];
#pragma unroll
    for (int v = 0; v < 8; v++) { mrow[v] = -1e30f; lrow[v] = 0.f; }

    for (int kb = 0; kb < S_; kb += 32) {
        // prefetch next K tile rows into cache hierarchy
        if (kb + 32 < S_) {
            __builtin_prefetch(Km + (size_t)(kb + 32 + l16) * C_ + h * HD_, 0, 3);
            __builtin_prefetch(vt0 + kb + 32, 0, 3);
        }

        // K^T B-fragments: [d=32][key=16] x2, straight from global [S,C]
        v8u k0u, k1u;
#pragma unroll
        for (int v = 0; v < 8; v++) {
            int d = a_kidx(v, hl);
            k0u[v] = *(const unsigned int*)(Km + (size_t)(kb + l16) * C_ + h * HD_ + d);
            k1u[v] = *(const unsigned int*)(Km + (size_t)(kb + 16 + l16) * C_ + h * HD_ + d);
        }
        v8f zero = {};
        v8f s0 = __builtin_amdgcn_wmma_f32_16x16x32_f16(false, qf, false, u8_to_h16(k0u),
                                                        (short)0, zero, false, false);
        v8f s1 = __builtin_amdgcn_wmma_f32_16x16x32_f16(false, qf, false, u8_to_h16(k1u),
                                                        (short)0, zero, false, false);
        // scale + relative-position bias
        int q0 = qbase + hl * 8;
#pragma unroll
        for (int v = 0; v < 8; v++) {
            int qq = q0 + v;
            int r0 = qq - (kb + l16) + (S_ - 1);
            s0[v] = s0[v] * scale + tab[(size_t)r0 * NH_ + h];
            s1[v] = s1[v] * scale + tab[(size_t)(r0 - 16) * NH_ + h];
        }
        // online softmax (row stats per 16-lane half)
#pragma unroll
        for (int v = 0; v < 8; v++) {
            float mc = red_max16(fmaxf(s0[v], s1[v]));
            float mn = fmaxf(mrow[v], mc);
            float corr = __expf(mrow[v] - mn);
            float p0 = __expf(s0[v] - mn);
            float p1 = __expf(s1[v] - mn);
            float rs = red_sum16(p0 + p1);
            lrow[v] = lrow[v] * corr + rs;
            mrow[v] = mn;
            o0[v] *= corr; o1[v] *= corr;
            s0[v] = p0; s1[v] = p1;
        }
        // D-layout -> A-layout for P via per-wave LDS tile
        _Float16* pp = sP[wid];
#pragma unroll
        for (int v = 0; v < 8; v++) {
            int row = v + hl * 8;
            pp[row * 32 + l16]      = (_Float16)s0[v];
            pp[row * 32 + 16 + l16] = (_Float16)s1[v];
        }
        asm volatile("s_wait_dscnt 0" ::: "memory");  // per-wave LDS RAW fence
        v8u pu;
#pragma unroll
        for (int v = 0; v < 8; v++)
            pu[v] = *(const unsigned int*)(pp + l16 * 32 + a_kidx(v, hl));
        v16h pf = u8_to_h16(pu);

        // V B-fragments [k=32][d=16] x2 from transposed V: contiguous u32 per VGPR
        v8u v0u, v1u;
#pragma unroll
        for (int v = 0; v < 8; v++) {
            int kk = a_kidx(v, hl);
            v0u[v] = *(const unsigned int*)(vt0 + kb + kk);
            v1u[v] = *(const unsigned int*)(vt1 + kb + kk);
        }
        o0 = __builtin_amdgcn_wmma_f32_16x16x32_f16(false, pf, false, u8_to_h16(v0u),
                                                    (short)0, o0, false, false);
        o1 = __builtin_amdgcn_wmma_f32_16x16x32_f16(false, pf, false, u8_to_h16(v1u),
                                                    (short)0, o1, false, false);
    }
    // finalize
#pragma unroll
    for (int v = 0; v < 8; v++) {
        float inv = 1.f / lrow[v];
        int row = qbase + v + hl * 8;
        out[(size_t)row * C_ + h * HD_ + l16]      = (_Float16)(o0[v] * inv);
        out[(size_t)row * C_ + h * HD_ + 16 + l16] = (_Float16)(o1[v] * inv);
    }
}

// ---------------------------------------------------------------- cross-attention (Lk=77)
// one wave per (query, head); 8 waves / block.
__global__ __launch_bounds__(256) void k_cross_attn(const _Float16* __restrict__ Q,
                                                    const _Float16* __restrict__ Kc,
                                                    const _Float16* __restrict__ Vc,
                                                    const float* __restrict__ tab,
                                                    _Float16* __restrict__ out) {
    __shared__ float sp[8][96];
    int lane = threadIdx.x & 31, wid = threadIdx.x >> 5;
    int task = blockIdx.x * 8 + wid;
    int q = task >> 3, h = task & 7;
    const float scale = 0.17677669529663687f;
    const _Float16* qp = Q + (size_t)q * C_ + h * HD_;

    float sc[3]; float mx = -1e30f;
#pragma unroll
    for (int j = 0; j < 3; j++) {
        int key = lane + 32 * j;
        float s = -1e30f;
        if (key < LC_) {
            const _Float16* kp = Kc + (size_t)key * C_ + h * HD_;
            float acc = 0.f;
#pragma unroll
            for (int d = 0; d < HD_; d++) acc += (float)qp[d] * (float)kp[d];
            s = acc * scale + tab[(size_t)(q - key + LC_ - 1) * NH_ + h];
        }
        sc[j] = s; mx = fmaxf(mx, s);
    }
    mx = red_max32(mx);
    float sum = 0.f;
    float e[3];
#pragma unroll
    for (int j = 0; j < 3; j++) {
        int key = lane + 32 * j;
        e[j] = (key < LC_) ? __expf(sc[j] - mx) : 0.f;
        sum += e[j];
    }
    sum = red_sum32(sum);
#pragma unroll
    for (int j = 0; j < 3; j++) {
        int key = lane + 32 * j;
        if (key < 96) sp[wid][key] = e[j];
    }
    asm volatile("s_wait_dscnt 0" ::: "memory");
    // lane = output head-dim
    float acc = 0.f;
    for (int key = 0; key < LC_; key++)
        acc += sp[wid][key] * (float)Vc[(size_t)key * C_ + h * HD_ + lane];
    out[(size_t)q * C_ + h * HD_ + lane] = (_Float16)(acc / sum);
}

// ---------------------------------------------------------------- host side
static inline void launch_gemm(const _Float16* A, const _Float16* B, const float* bias,
                               const float* resid, float* outF, _Float16* outH, _Float16* outHT,
                               int M, int N, int K, int gelu, hipStream_t s) {
    dim3 grid(N / 64, (M + 63) / 64);
    k_gemm<<<grid, 128, 0, s>>>(A, B, bias, resid, outF, outH, outHT, M, N, K, gelu);
}

extern "C" void kernel_launch(void* const* d_in, const int* in_sizes, int n_in,
                              void* d_out, int out_size, void* d_ws, size_t ws_size,
                              hipStream_t stream) {
    const float* hid    = (const float*)d_in[0];
    const float* ctx    = (const float*)d_in[1];
    const float* cn_g   = (const float*)d_in[2];
    const float* cn_b   = (const float*)d_in[3];
    const float* ca_wq  = (const float*)d_in[4];
    const float* ca_wk  = (const float*)d_in[5];
    const float* ca_wv  = (const float*)d_in[6];
    const float* ca_wo  = (const float*)d_in[7];
    const float* ca_bo  = (const float*)d_in[8];
    const float* ca_tab = (const float*)d_in[9];
    const float* sn_g   = (const float*)d_in[10];
    const float* sn_b   = (const float*)d_in[11];
    const float* sa_wq  = (const float*)d_in[12];
    const float* sa_wk  = (const float*)d_in[13];
    const float* sa_wv  = (const float*)d_in[14];
    const float* sa_wo  = (const float*)d_in[15];
    const float* sa_bo  = (const float*)d_in[16];
    const float* sa_tab = (const float*)d_in[17];
    const float* fn_g   = (const float*)d_in[18];
    const float* fn_b   = (const float*)d_in[19];
    const float* ff_w1  = (const float*)d_in[20];
    const float* ff_b1  = (const float*)d_in[21];
    const float* ff_w2  = (const float*)d_in[22];
    const float* ff_b2  = (const float*)d_in[23];

    size_t off = 0;
    auto alloc = [&](size_t bytes) {
        void* p = (char*)d_ws + off;
        off += (bytes + 255) & ~(size_t)255;
        return p;
    };
    float*    X    = (float*)   alloc((size_t)S_ * C_ * 4);
    _Float16* XN   = (_Float16*)alloc((size_t)S_ * C_ * 2);
    _Float16* QH   = (_Float16*)alloc((size_t)S_ * C_ * 2);
    _Float16* KH   = (_Float16*)alloc((size_t)S_ * C_ * 2);
    _Float16* VH   = (_Float16*)alloc((size_t)S_ * C_ * 2);   // cross V (row) / self V^T
    _Float16* AH   = (_Float16*)alloc((size_t)S_ * C_ * 2);
    _Float16* FF1H = (_Float16*)alloc((size_t)S_ * FF_ * 2);
    _Float16* CTXH = (_Float16*)alloc((size_t)LC_ * DC_ * 2);
    _Float16* WBUF = (_Float16*)alloc((size_t)C_ * FF_ * 2);   // max weight tile
    (void)ws_size; (void)in_sizes; (void)n_in; (void)out_size;

    const int NTOK = S_ * C_;
    k_transpose_in<<<(NTOK + 255) / 256, 256, 0, stream>>>(hid, X);
    k_f32_to_f16<<<256, 256, 0, stream>>>(ctx, CTXH, LC_ * DC_);

    for (int i = 0; i < NLAYER; i++) {
        // ---------------- cross-attention block ----------------
        k_layernorm<<<S_ / 8, 256, 0, stream>>>(X, cn_g + i * C_, cn_b + i * C_, XN, S_);
        k_f32_to_f16<<<256, 256, 0, stream>>>(ca_wq + (size_t)i * C_ * C_, WBUF, C_ * C_);
        launch_gemm(XN, WBUF, nullptr, nullptr, nullptr, QH, nullptr, S_, C_, C_, 0, stream);
        k_f32_to_f16<<<256, 256, 0, stream>>>(ca_wk + (size_t)i * DC_ * C_, WBUF, DC_ * C_);
        launch_gemm(CTXH, WBUF, nullptr, nullptr, nullptr, KH, nullptr, LC_, C_, DC_, 0, stream);
        k_f32_to_f16<<<256, 256, 0, stream>>>(ca_wv + (size_t)i * DC_ * C_, WBUF, DC_ * C_);
        launch_gemm(CTXH, WBUF, nullptr, nullptr, nullptr, VH, nullptr, LC_, C_, DC_, 0, stream);
        k_cross_attn<<<S_ * NH_ / 8, 256, 0, stream>>>(
            QH, KH, VH, ca_tab + (size_t)i * (S_ + LC_ - 1) * NH_, AH);
        k_f32_to_f16<<<256, 256, 0, stream>>>(ca_wo + (size_t)i * C_ * C_, WBUF, C_ * C_);
        launch_gemm(AH, WBUF, ca_bo + i * C_, X, X, nullptr, nullptr, S_, C_, C_, 0, stream);

        // ---------------- self-attention block ----------------
        k_layernorm<<<S_ / 8, 256, 0, stream>>>(X, sn_g + i * C_, sn_b + i * C_, XN, S_);
        k_f32_to_f16<<<256, 256, 0, stream>>>(sa_wq + (size_t)i * C_ * C_, WBUF, C_ * C_);
        launch_gemm(XN, WBUF, nullptr, nullptr, nullptr, QH, nullptr, S_, C_, C_, 0, stream);
        k_f32_to_f16<<<256, 256, 0, stream>>>(sa_wk + (size_t)i * C_ * C_, WBUF, C_ * C_);
        launch_gemm(XN, WBUF, nullptr, nullptr, nullptr, KH, nullptr, S_, C_, C_, 0, stream);
        k_f32_to_f16<<<256, 256, 0, stream>>>(sa_wv + (size_t)i * C_ * C_, WBUF, C_ * C_);
        launch_gemm(XN, WBUF, nullptr, nullptr, nullptr, nullptr, VH /*V^T*/, S_, C_, C_, 0, stream);
        k_flash_attn<<<dim3(S_ / 64, NH_), 128, 0, stream>>>(
            QH, KH, VH, sa_tab + (size_t)i * (2 * S_ - 1) * NH_, AH);
        k_f32_to_f16<<<256, 256, 0, stream>>>(sa_wo + (size_t)i * C_ * C_, WBUF, C_ * C_);
        launch_gemm(AH, WBUF, sa_bo + i * C_, X, X, nullptr, nullptr, S_, C_, C_, 0, stream);

        // ---------------- FFN (replaces x, no residual) ----------------
        k_layernorm<<<S_ / 8, 256, 0, stream>>>(X, fn_g + i * C_, fn_b + i * C_, XN, S_);
        k_f32_to_f16<<<256, 256, 0, stream>>>(ff_w1 + (size_t)i * C_ * FF_, WBUF, C_ * FF_);
        launch_gemm(XN, WBUF, ff_b1 + i * FF_, nullptr, nullptr, FF1H, nullptr, S_, FF_, C_, 1, stream);
        k_f32_to_f16<<<256, 256, 0, stream>>>(ff_w2 + (size_t)i * FF_ * C_, WBUF, FF_ * C_);
        launch_gemm(FF1H, WBUF, ff_b2 + i * C_, nullptr, X, nullptr, nullptr, S_, C_, FF_, 0, stream);
    }

    k_transpose_out<<<(NTOK + 255) / 256, 256, 0, stream>>>(X, (float*)d_out);
}
